// BAGenerator_attention_16724602650910
// MI455X (gfx1250) — matrix operand order repaired and checked by reading, not compiled, not run on hardware
//
#include <hip/hip_runtime.h>
#include <hip/hip_bf16.h>
#include <math.h>

// ---------------------------------------------------------------------------
// Problem constants (from reference)
// ---------------------------------------------------------------------------
#define NN    2048   // batch
#define TT    81     // frames
#define JJ    16     // joints
#define NBb   15     // bones
#define DA    48     // attention dim
#define HH    6      // heads
#define SEQQ  31     // sequence
#define DL    256    // hidden dim
#define KFC1  1536   // 1488 padded to 48*32
#define M2    (NN * 32)          // row-padded M for per-seq GEMMs
#define RSQ_BN 0.99999500003749981f   // 1/sqrt(1+1e-5)
#define INV_SQRT8 0.35355339059327373f

#define ACT_NONE     0
#define ACT_BN_LEAKY 1
#define ACT_LEAKY    2

typedef __attribute__((ext_vector_type(16))) _Float16 v16h;
typedef __attribute__((ext_vector_type(8)))  _Float16 v8h;
typedef __attribute__((ext_vector_type(8)))  float    v8f;

// Kinematic chain: bone b has child joint b+1; parent joint table:
__constant__ int c_BP[NBb] = {0,1,2,0,4,5,0,7,8,9,8,11,12,8,14};
// Per-joint bitmask of bones on path to root (M_BONE2POSE as bitmask):
__constant__ unsigned c_JMASK[JJ] = {
  0x0000,0x0001,0x0003,0x0007,0x0008,0x0018,0x0038,0x0040,
  0x00C0,0x01C0,0x03C0,0x04C0,0x0CC0,0x1CC0,0x20C0,0x60C0};

// Packed-B fragment sizes (halfs): Ntiles * (Kp/32) * 32 lanes * 16 halfs
#define QKV_KP      64
#define QKV_NTILES  4            // 48 cols padded to 64
#define QKV_FRAG    (QKV_NTILES * (QKV_KP/32) * 32 * 16)   // 4096
#define W1_NTILES   16
#define W1_FRAG     (W1_NTILES * (KFC1/32) * 32 * 16)      // 393216
#define W2_KP       256
#define W2_NTILES   4
#define W2_FRAG     (W2_NTILES * (W2_KP/32) * 32 * 16)     // 16384

// ---------------------------------------------------------------------------
// 1) Weight repack: f32 row-major W[j][k] -> f16 *WMMA B-fragment order*:
//    D[((tn*(Kp/32)+kb)*32 + lane)*16 + e]  holds  B[k][j] = W[j][k]  with
//    k = kb*32 + (e<8 ? (lane>>4)*8 + e : 16 + (lane>>4)*8 + e-8),  j = tn*16 + (lane&15); zero padded.
//    At GEMM time each lane loads its whole fragment as one 32B vector.
// ---------------------------------------------------------------------------
__device__ inline void pack_frag_elem(const float* __restrict__ W, int Ksrc, int Nsrc,
                                      _Float16* __restrict__ D, int Kp, int idx)
{
  const int e    = idx & 15;
  const int lane = (idx >> 4) & 31;
  const int rest = idx >> 9;
  const int kb32 = Kp >> 5;
  const int kb   = rest % kb32;
  const int tn   = rest / kb32;
  const int k    = kb * 32 + (e < 8 ? (lane >> 4) * 8 + e : 16 + (lane >> 4) * 8 + (e - 8));
  const int j    = tn * 16 + (lane & 15);
  float v = 0.0f;
  if (k < Ksrc && j < Nsrc) v = W[(size_t)j * Ksrc + k];
  D[idx] = (_Float16)v;
}

__global__ void pack_kernel(const float* __restrict__ wq, const float* __restrict__ wk,
                            const float* __restrict__ wv, const float* __restrict__ wo,
                            const float* __restrict__ w1, const float* __restrict__ w2,
                            const float* __restrict__ b2,
                            _Float16* __restrict__ WqB, _Float16* __restrict__ WkB,
                            _Float16* __restrict__ WvB, _Float16* __restrict__ WoB,
                            _Float16* __restrict__ W1B, _Float16* __restrict__ W2B,
                            float* __restrict__ b2p)
{
  const int total = 4 * QKV_FRAG + W1_FRAG + W2_FRAG + 64;
  for (int i = blockIdx.x * blockDim.x + threadIdx.x; i < total;
       i += gridDim.x * blockDim.x) {
    int idx = i;
    if (idx < 4 * QKV_FRAG) {
      int w = idx / QKV_FRAG; int r = idx % QKV_FRAG;
      const float* W = (w == 0) ? wq : (w == 1) ? wk : (w == 2) ? wv : wo;
      _Float16*    D = (w == 0) ? WqB : (w == 1) ? WkB : (w == 2) ? WvB : WoB;
      pack_frag_elem(W, DA, DA, D, QKV_KP, r);
      continue;
    }
    idx -= 4 * QKV_FRAG;
    if (idx < W1_FRAG) { pack_frag_elem(w1, SEQQ * DA, DL, W1B, KFC1, idx); continue; }
    idx -= W1_FRAG;
    if (idx < W2_FRAG) { pack_frag_elem(w2, DL, NBb * 4, W2B, W2_KP, idx); continue; }
    idx -= W2_FRAG;
    b2p[idx] = (idx < NBb * 4) ? b2[idx] : 0.0f;                 // padded b2 (64)
  }
}

// ---------------------------------------------------------------------------
// 2) Encoder + BN + ReLU + LayerNorm per batch element.
//    Yenc: f32 (N*32 x 48), row 31 zeroed (residual stream).
//    Hp:   f16 (N*32 x 64), padded with zeros (A-matrix for QKV GEMMs).
// ---------------------------------------------------------------------------
__global__ __launch_bounds__(64)
void encode_kernel(const float* __restrict__ in3d, const float* __restrict__ noise,
                   const float* __restrict__ enc_w, const float* __restrict__ enc_b,
                   const float* __restrict__ g, const float* __restrict__ bb,
                   const float* __restrict__ ln_a, const float* __restrict__ ln_b,
                   float* __restrict__ Yenc, _Float16* __restrict__ Hp)
{
  const int n = blockIdx.x, tid = threadIdx.x;
  __shared__ float yin[SEQQ * 3];        // 93 = 48 mid-frame coords + 45 noise
  __shared__ float y0[SEQQ][DA];
  const int mid = (TT - 1) / 2;
  const size_t base = ((size_t)n * TT + mid) * (JJ * 3);

  for (int i = tid; i < SEQQ * 3; i += 64) {
    float v;
    if (i < JJ * 3) { int j = i / 3, c = i % 3;
      v = in3d[base + j * 3 + c] - in3d[base + c];     // center on root joint
    } else v = noise[(size_t)n * 45 + (i - JJ * 3)];
    yin[i] = v;
  }
  __syncthreads();

  for (int i = tid; i < SEQQ * DA; i += 64) {
    int s = i / DA, c = i % DA;
    float v = enc_b[c]
            + yin[s*3+0] * enc_w[c*3+0]
            + yin[s*3+1] * enc_w[c*3+1]
            + yin[s*3+2] * enc_w[c*3+2];
    v = v * g[c] * RSQ_BN + bb[c];
    v = v > 0.0f ? v : 0.0f;
    y0[s][c] = v;
    Yenc[((size_t)n * 32 + s) * DA + c] = v;
  }
  for (int i = tid; i < DA; i += 64)            // zero pad row 31
    Yenc[((size_t)n * 32 + 31) * DA + i] = 0.0f;
  __syncthreads();

  for (int s = tid; s < 32; s += 64) {
    _Float16* Hrow = Hp + ((size_t)n * 32 + s) * 64;
    if (s < SEQQ) {
      float mean = 0.0f;
      for (int c = 0; c < DA; ++c) mean += y0[s][c];
      mean *= (1.0f / DA);
      float var = 0.0f;
      for (int c = 0; c < DA; ++c) { float d = y0[s][c] - mean; var += d * d; }
      float stdv = sqrtf(var * (1.0f / (DA - 1)));
      float inv  = 1.0f / (stdv + 1e-6f);
      for (int c = 0; c < DA; ++c)
        Hrow[c] = (_Float16)(ln_a[c] * (y0[s][c] - mean) * inv + ln_b[c]);
      for (int c = DA; c < 64; ++c) Hrow[c] = (_Float16)0.0f;
    } else {
      for (int c = 0; c < 64; ++c) Hrow[c] = (_Float16)0.0f;
    }
  }
}

// ---------------------------------------------------------------------------
// 3) WMMA GEMM: one wave32 computes TWO adjacent 16x16 C tiles (shared A
//    fragment), K unrolled x2 -> 4 v_wmma per loop body.
//    A: row-major f16 (lda mult of 32); per-lane fragment = two aligned b128
//       loads at k0+grp*8 and k0+16+grp*8 (CDNA5 16-bit A layout).
//    Bp: fragment-packed (see pack_kernel); per-lane fragment = one 32B load.
// ---------------------------------------------------------------------------
__device__ inline v16h load_a_frag(const _Float16* __restrict__ Arow, int k0, int grp)
{
  v8h lo = *(const v8h*)(Arow + k0 + grp * 8);        // K = grp*8 .. +7
  v8h hi = *(const v8h*)(Arow + k0 + 16 + grp * 8);   // K = 16+grp*8 .. +7
  return __builtin_shufflevector(lo, hi, 0,1,2,3,4,5,6,7,8,9,10,11,12,13,14,15);
}

__device__ inline void epilogue(v8f acc, int tm, int tn, int grp, int lidx,
                                const float* __restrict__ bias,
                                const float* __restrict__ resid, int ldr,
                                float* __restrict__ Cf, _Float16* __restrict__ Ch,
                                int ldc, const float* __restrict__ bn_g,
                                const float* __restrict__ bn_b, int Nstore, int act)
{
  const int col = tn * 16 + lidx;
  if (col >= Nstore) return;
  const float bsv = bias ? bias[col] : 0.0f;
#pragma unroll
  for (int r = 0; r < 8; ++r) {
    int row = tm * 16 + grp * 8 + r;          // C layout: VGPR r -> M=r / r+8
    float v = acc[r] + bsv;
    if (resid) v += resid[(size_t)row * ldr + col];
    if (act == ACT_BN_LEAKY) {
      v = v * bn_g[col] * RSQ_BN + bn_b[col];
      v = v >= 0.0f ? v : 0.01f * v;
    } else if (act == ACT_LEAKY) {
      v = v >= 0.0f ? v : 0.01f * v;
    }
    if (Cf) Cf[(size_t)row * ldc + col] = v;
    if (Ch) Ch[(size_t)row * ldc + col] = (_Float16)v;
  }
}

__global__ __launch_bounds__(32)
void gemm_wmma(const _Float16* __restrict__ A, int lda,
               const _Float16* __restrict__ Bp,
               const float* __restrict__ bias,
               const float* __restrict__ resid, int ldr,
               float* __restrict__ Cf, _Float16* __restrict__ Ch, int ldc,
               const float* __restrict__ bn_g, const float* __restrict__ bn_b,
               int tilePairsN, int Nstore, int K, int act)
{
  const int tile = blockIdx.x;
  const int tm = tile / tilePairsN;
  const int tp = tile - tm * tilePairsN;
  const int tn0 = tp * 2, tn1 = tn0 + 1;
  const int lane = threadIdx.x;
  const int grp  = lane >> 4;       // 0: lanes 0-15, 1: lanes 16-31
  const int lidx = lane & 15;
  const int kb32 = K >> 5;          // # of 32-wide K blocks (always even here)

  const _Float16* Arow = A + (size_t)(tm * 16 + lidx) * lda;
  const _Float16* B0 = Bp + (((size_t)tn0 * kb32) * 32 + lane) * 16;
  const _Float16* B1 = Bp + (((size_t)tn1 * kb32) * 32 + lane) * 16;

  v8f acc0 = {}, acc1 = {};
  for (int kb = 0; kb < kb32; kb += 2) {
    // ---- K block kb
    {
      v16h a  = load_a_frag(Arow, kb * 32, grp);
      v16h b0 = *(const v16h*)(B0 + (size_t)kb * (32 * 16));
      v16h b1 = *(const v16h*)(B1 + (size_t)kb * (32 * 16));
      acc0 = __builtin_amdgcn_wmma_f32_16x16x32_f16(false, a, false, b0,
                                                    (short)0, acc0, false, false);
      acc1 = __builtin_amdgcn_wmma_f32_16x16x32_f16(false, a, false, b1,
                                                    (short)0, acc1, false, false);
    }
    // ---- K block kb+1
    {
      v16h a  = load_a_frag(Arow, (kb + 1) * 32, grp);
      v16h b0 = *(const v16h*)(B0 + (size_t)(kb + 1) * (32 * 16));
      v16h b1 = *(const v16h*)(B1 + (size_t)(kb + 1) * (32 * 16));
      acc0 = __builtin_amdgcn_wmma_f32_16x16x32_f16(false, a, false, b0,
                                                    (short)0, acc0, false, false);
      acc1 = __builtin_amdgcn_wmma_f32_16x16x32_f16(false, a, false, b1,
                                                    (short)0, acc1, false, false);
    }
  }

  epilogue(acc0, tm, tn0, grp, lidx, bias, resid, ldr, Cf, Ch, ldc,
           bn_g, bn_b, Nstore, act);
  epilogue(acc1, tm, tn1, grp, lidx, bias, resid, ldr, Cf, Ch, ldc,
           bn_g, bn_b, Nstore, act);
}

// ---------------------------------------------------------------------------
// 4) Softmax attention per (n, head): 31 queries x 31 keys, head dim 8.
//    Writes padded f16 context (N*32 x 64) for the WMMA output projection.
// ---------------------------------------------------------------------------
__global__ __launch_bounds__(32)
void attention_kernel(const float* __restrict__ Qf, const float* __restrict__ Kf,
                      const float* __restrict__ Vf, _Float16* __restrict__ CtxH)
{
  const int nh = blockIdx.x;
  const int n = nh / HH, h = nh % HH;
  const int s = threadIdx.x;
  __shared__ float Ks[SEQQ][8];
  __shared__ float Vs[SEQQ][8];

  for (int i = s; i < SEQQ * 8; i += 32) {
    int r = i / 8, d = i % 8;
    Ks[r][d] = Kf[((size_t)n * 32 + r) * DA + h * 8 + d];
    Vs[r][d] = Vf[((size_t)n * 32 + r) * DA + h * 8 + d];
  }
  __syncthreads();

  if (s < SEQQ) {
    float q[8];
#pragma unroll
    for (int d = 0; d < 8; ++d) q[d] = Qf[((size_t)n * 32 + s) * DA + h * 8 + d];
    float sc[SEQQ];
    float mx = -1e30f;
    for (int k = 0; k < SEQQ; ++k) {
      float t = 0.0f;
#pragma unroll
      for (int d = 0; d < 8; ++d) t += q[d] * Ks[k][d];
      t *= INV_SQRT8;
      sc[k] = t; mx = t > mx ? t : mx;
    }
    float sum = 0.0f;
    for (int k = 0; k < SEQQ; ++k) { float e = expf(sc[k] - mx); sc[k] = e; sum += e; }
    float inv = 1.0f / sum;
    _Float16* Crow = CtxH + ((size_t)n * 32 + s) * 64;
#pragma unroll
    for (int d = 0; d < 8; ++d) {
      float a = 0.0f;
      for (int k = 0; k < SEQQ; ++k) a += sc[k] * Vs[k][d];
      Crow[h * 8 + d] = (_Float16)(a * inv);
    }
    if (h == 0) for (int c = DA; c < 64; ++c) Crow[c] = (_Float16)0.0f;
  } else if (h == 0) {                           // s == 31: zero padded row
    _Float16* Crow = CtxH + ((size_t)n * 32 + 31) * 64;
    for (int c = 0; c < 64; ++c) Crow[c] = (_Float16)0.0f;
  }
}

// ---------------------------------------------------------------------------
// 5) Flatten residual stream (N*32 x 48 f32) -> FC1 input (N x 1536 f16).
// ---------------------------------------------------------------------------
__global__ __launch_bounds__(256)
void fc1_input_kernel(const float* __restrict__ Yres, _Float16* __restrict__ Yflat)
{
  const int n = blockIdx.x;
  for (int i = threadIdx.x; i < KFC1; i += 256) {
    float v = 0.0f;
    if (i < SEQQ * DA) { int s = i / DA, c = i % DA;
      v = Yres[((size_t)n * 32 + s) * DA + c]; }
    Yflat[(size_t)n * KFC1 + i] = (_Float16)v;
  }
}

// ---------------------------------------------------------------------------
// 6) Per-(n,t) finalize: bone vectors, Rodrigues rotation, fixed-bone mask,
//    chain accumulation via ancestor bitmasks, write pose + ba_diff.
// ---------------------------------------------------------------------------
__global__ __launch_bounds__(32)
void finalize_kernel(const float* __restrict__ in3d, const float* __restrict__ params,
                     float* __restrict__ out, float* __restrict__ badiff)
{
  const int bt = blockIdx.x;
  const int n = bt / TT, t = bt % TT;
  const int lane = threadIdx.x;
  __shared__ float xj[JJ][3];
  __shared__ float mb[NBb][3];
  const size_t fbase = ((size_t)n * TT + t) * (JJ * 3);

  if (lane < JJ) {
#pragma unroll
    for (int c = 0; c < 3; ++c) xj[lane][c] = in3d[fbase + lane * 3 + c];
  }
  __syncthreads();

  if (lane < NBb) {
    const int pj = c_BP[lane], cj = lane + 1;
    float bx = xj[cj][0] - xj[pj][0];
    float by = xj[cj][1] - xj[pj][1];
    float bz = xj[cj][2] - xj[pj][2];
    float bl = sqrtf(bx*bx + by*by + bz*bz);
    float ux = bx / bl, uy = by / bl, uz = bz / bl;

    const float* p = params + (size_t)n * 64 + lane * 4;
    float a0 = p[0], a1 = p[1], a2 = p[2], th = p[3];
    float inva = 1.0f / sqrtf(a0*a0 + a1*a1 + a2*a2);
    a0 *= inva; a1 *= inva; a2 *= inva;
    float ang = (th * (1.0f / (float)TT)) * (float)t;
    float v0 = a0 * ang, v1 = a1 * ang, v2 = a2 * ang;
    float nv = sqrtf(v0*v0 + v1*v1 + v2*v2);
    float invn = 1.0f / fmaxf(nv, 1e-8f);
    float k0 = v0 * invn, k1 = v1 * invn, k2 = v2 * invn;
    float cc = cosf(nv), ss = sinf(nv);
    float kdu = k0*ux + k1*uy + k2*uz;
    float cx = k1*uz - k2*uy, cy = k2*ux - k0*uz, cz = k0*uy - k1*ux;
    float r0 = cc*ux + ss*cx + (1.0f - cc) * k0 * kdu;
    float r1 = cc*uy + ss*cy + (1.0f - cc) * k1 * kdu;
    float r2 = cc*uz + ss*cz + (1.0f - cc) * k2 * kdu;
    if (lane == 6 || lane == 7) { r0 = ux; r1 = uy; r2 = uz; }   // FIXED_MASK
    badiff[(size_t)bt * NBb + lane] = 1.0f - (r0*ux + r1*uy + r2*uz);
    mb[lane][0] = r0 * bl; mb[lane][1] = r1 * bl; mb[lane][2] = r2 * bl;
  }
  __syncthreads();

  if (lane < JJ) {
    const unsigned m = c_JMASK[lane];
    float o0 = xj[0][0], o1 = xj[0][1], o2 = xj[0][2];    // + root
    for (int b = 0; b < NBb; ++b)
      if ((m >> b) & 1u) { o0 += mb[b][0]; o1 += mb[b][1]; o2 += mb[b][2]; }
    out[fbase + lane * 3 + 0] = o0;
    out[fbase + lane * 3 + 1] = o1;
    out[fbase + lane * 3 + 2] = o2;
  }
}

// ---------------------------------------------------------------------------
// Launcher
// ---------------------------------------------------------------------------
extern "C" void kernel_launch(void* const* d_in, const int* in_sizes, int n_in,
                              void* d_out, int out_size, void* d_ws, size_t ws_size,
                              hipStream_t stream) {
  const float* in3d   = (const float*)d_in[0];
  const float* noise  = (const float*)d_in[1];
  const float* enc_w  = (const float*)d_in[2];
  const float* enc_b  = (const float*)d_in[3];
  const float* ebn_g  = (const float*)d_in[4];
  const float* ebn_b  = (const float*)d_in[5];
  const float* ln_a   = (const float*)d_in[6];
  const float* ln_b   = (const float*)d_in[7];
  const float* wq     = (const float*)d_in[8];
  const float* bq     = (const float*)d_in[9];
  const float* wk     = (const float*)d_in[10];
  const float* bk     = (const float*)d_in[11];
  const float* wv     = (const float*)d_in[12];
  const float* bv     = (const float*)d_in[13];
  const float* wo     = (const float*)d_in[14];
  const float* bo     = (const float*)d_in[15];
  const float* w1     = (const float*)d_in[16];
  const float* b1     = (const float*)d_in[17];
  const float* bn1_g  = (const float*)d_in[18];
  const float* bn1_b  = (const float*)d_in[19];
  const float* w2     = (const float*)d_in[20];
  const float* b2     = (const float*)d_in[21];

  float* out    = (float*)d_out;
  float* badiff = out + (size_t)NN * TT * JJ * 3;

  // ---- workspace bump allocator (256B aligned), with aliasing of dead bufs
  char* ws = (char*)d_ws;
  size_t o = 0;
  auto take = [&](size_t bytes) -> char* {
    char* p = ws + o;
    o = (o + bytes + 255) & ~(size_t)255;
    return p;
  };
  _Float16* W1B  = (_Float16*)take((size_t)W1_FRAG * 2);
  _Float16* W2B  = (_Float16*)take((size_t)W2_FRAG * 2);
  _Float16* WqB  = (_Float16*)take((size_t)QKV_FRAG * 2);
  _Float16* WkB  = (_Float16*)take((size_t)QKV_FRAG * 2);
  _Float16* WvB  = (_Float16*)take((size_t)QKV_FRAG * 2);
  _Float16* WoB  = (_Float16*)take((size_t)QKV_FRAG * 2);
  float*    b2p  = (float*)take(64 * 4);
  float*    Yenc = (float*)take((size_t)M2 * DA * 4);
  _Float16* Hp   = (_Float16*)take((size_t)M2 * 64 * 2);
  float*    Qf   = (float*)take((size_t)M2 * DA * 4);
  float*    Kf   = (float*)take((size_t)M2 * DA * 4);
  float*    Vf   = (float*)take((size_t)M2 * DA * 4);
  _Float16* CtxH = (_Float16*)take((size_t)M2 * 64 * 2);
  // aliases onto dead buffers (stream-ordered, same stream => safe):
  float*    Yres   = Qf;                 // written after Q/K/V fully consumed
  _Float16* Yflat  = (_Float16*)Kf;      // FC1 input, written after attention
  _Float16* Z1h    = (_Float16*)Vf;      // FC1 output
  float*    Params = (float*)Hp;         // FC2 output (Hp dead after QKV)

  // ---- 1) repack weights into WMMA B-fragment order
  pack_kernel<<<512, 256, 0, stream>>>(wq, wk, wv, wo, w1, w2, b2,
                                       WqB, WkB, WvB, WoB, W1B, W2B, b2p);
  // ---- 2) encoder + layernorm
  encode_kernel<<<NN, 64, 0, stream>>>(in3d, noise, enc_w, enc_b, ebn_g, ebn_b,
                                       ln_a, ln_b, Yenc, Hp);
  // ---- 3) Q/K/V projections: (N*32 x 64) @ (64 x 48pad64), WMMA, 2 N-tiles/wave
  const int qkvBlocks = (M2 / 16) * (QKV_NTILES / 2);   // 4096 * 2
  gemm_wmma<<<qkvBlocks, 32, 0, stream>>>(Hp, 64, WqB, bq, nullptr, 0,
                                          Qf, nullptr, DA, nullptr, nullptr,
                                          QKV_NTILES / 2, DA, QKV_KP, ACT_NONE);
  gemm_wmma<<<qkvBlocks, 32, 0, stream>>>(Hp, 64, WkB, bk, nullptr, 0,
                                          Kf, nullptr, DA, nullptr, nullptr,
                                          QKV_NTILES / 2, DA, QKV_KP, ACT_NONE);
  gemm_wmma<<<qkvBlocks, 32, 0, stream>>>(Hp, 64, WvB, bv, nullptr, 0,
                                          Vf, nullptr, DA, nullptr, nullptr,
                                          QKV_NTILES / 2, DA, QKV_KP, ACT_NONE);
  // ---- 4) softmax attention per (n, head)
  attention_kernel<<<NN * HH, 32, 0, stream>>>(Qf, Kf, Vf, CtxH);
  // ---- 5) output projection + residual: Yres = Yenc + Ctx @ wo.T + bo, WMMA
  gemm_wmma<<<qkvBlocks, 32, 0, stream>>>(CtxH, 64, WoB, bo, Yenc, DA,
                                          Yres, nullptr, DA, nullptr, nullptr,
                                          QKV_NTILES / 2, DA, QKV_KP, ACT_NONE);
  // ---- 6) flatten to FC1 input
  fc1_input_kernel<<<NN, 256, 0, stream>>>(Yres, Yflat);
  // ---- 7) FC1: (N x 1536) @ (1536 x 256) + BN + LeakyReLU -> f16, WMMA
  gemm_wmma<<<(NN / 16) * (W1_NTILES / 2), 32, 0, stream>>>(
      Yflat, KFC1, W1B, b1, nullptr, 0,
      nullptr, Z1h, DL, bn1_g, bn1_b, W1_NTILES / 2, DL, KFC1, ACT_BN_LEAKY);
  // ---- 8) FC2: (N x 256) @ (256 x 64) + LeakyReLU -> f32 params, WMMA
  gemm_wmma<<<(NN / 16) * (W2_NTILES / 2), 32, 0, stream>>>(
      Z1h, DL, W2B, b2p, nullptr, 0,
      Params, nullptr, 64, nullptr, nullptr, W2_NTILES / 2, 64, W2_KP, ACT_LEAKY);
  // ---- 9) per-(n,t) pose reconstruction
  finalize_kernel<<<NN * TT, 32, 0, stream>>>(in3d, Params, out, badiff);
}
